// RecurrentCapsuleNetwork_85555748537122
// MI455X (gfx1250) — compile-verified, matrix operand order checked by reading, not run on hardware
//
#include <hip/hip_runtime.h>
#include <hip/hip_bf16.h>
#include <stdint.h>

// ---------------------------------------------------------------------------
// RecurrentCapsuleNetwork for MI455X (gfx1250), wave32 + WMMA bf16 + TDM.
//   Phase 0: cast x -> bf16, pack [W_hh|W_ih] -> bf16 WMMA tiles, init.
//   Phase 1: persistent LSTM, 64 WGs x 256 threads.
//            - weights (160KB/WG) TDM-loaded into LDS once
//            - per step: TDM loads h_t panel (64KB) + prefetches x_{t+1}
//              panel (16KB, ping-pong) into LDS; s_wait_tensorcnt 1 overlaps
//              the x prefetch with compute (TDM ops complete in order)
//            - 8 waves x 40 v_wmma_f32_16x16x32_bf16 all from LDS
//            - gate math, h ping-pong via global/L2, grid barrier
//   Phase 2: head kernel: u = hsum @ W_cap^T + T*b_cap; squash; @ W_out^T.
// ---------------------------------------------------------------------------

typedef __attribute__((ext_vector_type(16))) __bf16 v16bf;
typedef __attribute__((ext_vector_type(8)))  float  v8f;
typedef __attribute__((ext_vector_type(4)))  unsigned int u32x4;
typedef __attribute__((ext_vector_type(8)))  int i32x8;
typedef __attribute__((ext_vector_type(4)))  int i32x4;

#if defined(__has_builtin)
#if __has_builtin(__builtin_amdgcn_tensor_load_to_lds) && \
    __has_builtin(__builtin_amdgcn_s_wait_tensorcnt)
#define HAVE_TDM 1
#endif
#endif
#ifndef HAVE_TDM
#define HAVE_TDM 0
#endif

namespace rc {
constexpr int kB = 32, kT = 2048, kIN = 256, kH = 1024, kD = 256, kOUT = 512;
constexpr int kKTOT = kH + kIN;        // 1280 fused K
constexpr int kNKT  = kKTOT / 32;      // 40 k-tiles of 32
constexpr int kNH   = 16;              // hidden cols per WG
constexpr int kNWG  = kH / kNH;        // 64 persistent workgroups

// workspace layout (bytes)
constexpr size_t offXbf   = 0;
constexpr size_t szXbf    = (size_t)kT * kB * kIN * 2;          // 32 MB
constexpr size_t offWpack = offXbf + szXbf;
constexpr size_t szWpack  = (size_t)kNWG * 4 * kNKT * 1024;     // 10 MB
constexpr size_t offHbuf  = offWpack + szWpack;
constexpr size_t szHbuf   = 2ull * kB * kH * 2;                 // ping-pong h
constexpr size_t offHsum  = offHbuf + szHbuf;
constexpr size_t szHsum   = (size_t)kB * kH * 4;
constexpr size_t offBar   = (offHsum + szHsum + 255) & ~(size_t)255;

// LDS layout (dynamic shared, base assumed at LDS offset 0)
constexpr int kLdsW     = 4 * kNKT * 1024;          // 160 KB weight slice
constexpr int kOffHp    = kLdsW;                    // h_t panel  (64 KB)
constexpr int kHpSz     = kB * kH * 2;
constexpr int kOffXp    = kOffHp + kHpSz;           // x panels   (2 x 16 KB)
constexpr int kXpSz     = kB * kIN * 2;
constexpr int kOffStage = kOffXp + 2 * kXpSz;       // gate stage (8 KB)
constexpr int kLdsStage = 4 * kB * kNH * 4;
constexpr int kLdsTot   = kOffStage + kLdsStage;    // 264 KB <= 320 KB
}
using namespace rc;

// ---------------------------------------------------------------- Phase 0 ---
__global__ void k_cast_x(const float* __restrict__ x, __bf16* __restrict__ xbf) {
  int idx = blockIdx.x * 256 + threadIdx.x;       // enumerates [t][b][i]
  int i  = idx % kIN;
  int bt = idx / kIN;
  int b  = bt % kB;
  int t  = bt / kB;
  xbf[idx] = (__bf16)x[((size_t)b * kT + t) * kIN + i];
}

// Pack fused weight [4H, 1280] into per-WG / per-gate / per-ktile 1KB tiles.
__global__ void k_pack_w(const float* __restrict__ Whh,
                         const float* __restrict__ Wih,
                         __bf16* __restrict__ wp) {
  int idx  = blockIdx.x * 256 + threadIdx.x;      // over NWG*4*NKT*512
  int e    = idx & 511;
  int tile = idx >> 9;
  int kt   = tile % kNKT;
  int gate = (tile / kNKT) & 3;
  int wg   = tile / (kNKT * 4);
  int lane = e >> 4;
  int j    = e & 15;
  int n    = lane & 15;
  int kk   = ((lane >> 4) << 4) + j;
  int k    = kt * 32 + kk;
  int row  = gate * kH + wg * kNH + n;
  float v  = (k < kH) ? Whh[(size_t)row * kH + k]
                      : Wih[(size_t)row * kIN + (k - kH)];
  wp[idx] = (__bf16)v;
}

__global__ void k_init(__bf16* __restrict__ hbuf, int* __restrict__ bar) {
  int idx = blockIdx.x * 256 + threadIdx.x;
  if (idx < 2 * kB * kH) hbuf[idx] = (__bf16)0.0f;
  if (idx == 0) *bar = 0;
}

// ---------------------------------------------------------------- Phase 1 ---
__device__ __forceinline__ float fsig(float v) {
  return 1.0f / (1.0f + __expf(-v));
}
__device__ __forceinline__ float ftanh(float v) {
  float e2 = __expf(2.0f * v);
  return (e2 - 1.0f) / (e2 + 1.0f);
}

extern __shared__ char smem[];

#if HAVE_TDM
// 1-D TDM copy: nu8 units of 8 bytes, global -> LDS byte offset lds_off.
// D# per ISA §8.3/8.4: group0 = {ctrl, lds_addr, gaddr_lo, gaddr_hi|type=2},
// group1 = data_size=8B, tensor_dim0=tile_dim0=nu8, tensor_dim1=1.
__device__ __forceinline__ void tdm_load_1d(unsigned lds_off, const void* gsrc,
                                            unsigned nu8) {
  unsigned long long ga = (unsigned long long)(uintptr_t)gsrc;
  u32x4 g0;
  g0[0] = 1u;                                       // count=1 (valid), user
  g0[1] = lds_off;                                  // lds_addr [63:32]
  g0[2] = (unsigned)(ga & 0xffffffffu);             // global_addr[31:0]
  g0[3] = (unsigned)((ga >> 32) & 0x01ffffffu)      // global_addr[56:32]
          | (2u << 30);                             // type=2 ("image")
  i32x8 g1;
  g1[0] = (int)(3u << 16);                          // data_size=3 -> 8B units
  g1[1] = (int)((nu8 & 0xffffu) << 16);             // tensor_dim0[15:0]
  g1[2] = (int)(((nu8 >> 16) & 0xffffu) | (1u << 16)); // dim0 hi | tensor_dim1=1
  g1[3] = (int)((nu8 & 0xffffu) << 16);             // tile_dim0 = nu8
  g1[4] = 0;                                        // tile_dim1/2 unused
  g1[5] = (int)nu8;                                 // tensor_dim0_stride lo
  g1[6] = 0;
  g1[7] = 0;
  i32x4 z4 = {0, 0, 0, 0};
#if __clang_major__ >= 23
  i32x8 z8 = {0, 0, 0, 0, 0, 0, 0, 0};
  __builtin_amdgcn_tensor_load_to_lds(g0, g1, z4, z4, z8, 0);
#else
  __builtin_amdgcn_tensor_load_to_lds(g0, g1, z4, z4, 0);
#endif
}
#else
__device__ __forceinline__ void lds_copy16(char* dst, const char* src,
                                           int bytes, int tid) {
  const float4* s = (const float4*)src;
  float4* d = (float4*)dst;
  for (int i = tid; i < bytes / 16; i += 256) d[i] = s[i];
}
#endif

__global__ __launch_bounds__(256)
void k_lstm(const __bf16* __restrict__ xbf, const __bf16* __restrict__ wp,
            const float* __restrict__ b_ih, const float* __restrict__ b_hh,
            __bf16* __restrict__ hbuf, float* __restrict__ hsum,
            int* __restrict__ bar) {
  float* stage = (float*)(smem + kOffStage);      // [gate][32][16] f32

  const int tid  = threadIdx.x;
  const int lane = tid & 31;
  const int wave = tid >> 5;                      // 8 waves
  const int wg   = blockIdx.x;                    // hidden band wg*16..+15
  const int mtile = wave & 1;                     // batch half (16 rows)
  const int gate  = wave >> 1;                    // i,f,g,o

  // ---- one-time: weight slice (160KB) + x_0 panel into LDS ----
#if HAVE_TDM
  if (wave == 0) {
    tdm_load_1d(0u, wp + (size_t)wg * (4 * kNKT * 512), (unsigned)(kLdsW / 8));
    tdm_load_1d((unsigned)kOffXp, xbf, (unsigned)(kXpSz / 8));
    __builtin_amdgcn_s_wait_tensorcnt(0);
  }
#else
  lds_copy16(smem, (const char*)(wp + (size_t)wg * (4 * kNKT * 512)), kLdsW, tid);
  lds_copy16(smem + kOffXp, (const char*)xbf, kXpSz, tid);
#endif
  __syncthreads();

  // ---- per-thread persistent cell/hsum state (2 cells of 32x16) ----
  const int cell0 = tid * 2, cell1 = cell0 + 1;
  float c0 = 0.f, c1 = 0.f, s0 = 0.f, s1 = 0.f;
  float bia[2][4];
  {
    const int nn[2] = {cell0 & 15, cell1 & 15};
    for (int j = 0; j < 2; ++j)
      for (int g = 0; g < 4; ++g) {
        int col = g * kH + wg * kNH + nn[j];
        bia[j][g] = b_ih[col] + b_hh[col];
      }
  }

  const int arow = mtile * 16 + (lane & 15);      // A-fragment row
  const int ksel = (lane >> 4) << 4;              // K half select: this code's k = 16h + s order
  const char* const ah = smem + kOffHp + arow * (kH * 2) + ksel * 2;
  const char* const bb = smem + gate * (kNKT * 1024) + lane * 32;

  for (int t = 0; t < kT; ++t) {
    const __bf16* hsrc = hbuf + (size_t)(t & 1) * kB * kH;
    __bf16*       hdst = hbuf + (size_t)((t + 1) & 1) * kB * kH;

    // ---- stage A panel: h_t now, x_{t+1} prefetch (in-order TDM) ----
#if HAVE_TDM
    if (wave == 0) {
      tdm_load_1d((unsigned)kOffHp, hsrc, (unsigned)(kHpSz / 8));
      const int tn = (t + 1) & (kT - 1);          // wraps at end (harmless)
      tdm_load_1d((unsigned)(kOffXp + ((t + 1) & 1) * kXpSz),
                  xbf + (size_t)tn * kB * kIN, (unsigned)(kXpSz / 8));
      __builtin_amdgcn_s_wait_tensorcnt(1);       // h_t (and x_t) complete;
    }                                             // x_{t+1} stays in flight
#else
    lds_copy16(smem + kOffHp, (const char*)hsrc, kHpSz, tid);
    lds_copy16(smem + kOffXp + (t & 1) * kXpSz,
               (const char*)(xbf + (size_t)t * kB * kIN), kXpSz, tid);
#endif
    __syncthreads();

    // ---- [h_t | x_t] @ W_slice : K=1280, 40 WMMAs, all operands in LDS ----
    const char* ax = smem + kOffXp + (t & 1) * kXpSz + arow * (kIN * 2) + ksel * 2;
    v8f acc = {};
#pragma unroll
    for (int kt = 0; kt < 32; ++kt) {             // h part (K=1024)
      v16bf a = *(const v16bf*)(ah + kt * 64);
      v16bf b = *(const v16bf*)(bb + kt * 1024);
      acc = __builtin_amdgcn_wmma_f32_16x16x32_bf16(
          false, a, false, b, (short)0, acc, false, false);
    }
#pragma unroll
    for (int kt = 0; kt < 8; ++kt) {              // x part (K=256)
      v16bf a = *(const v16bf*)(ax + kt * 64);
      v16bf b = *(const v16bf*)(bb + (32 + kt) * 1024);
      acc = __builtin_amdgcn_wmma_f32_16x16x32_bf16(
          false, a, false, b, (short)0, acc, false, false);
    }

    // ---- stash C tile: (gate, m, n) -> stage[gate*512 + m*16 + n] ----
    {
      const int nn = lane & 15;
#pragma unroll
      for (int r = 0; r < 8; ++r) {
        int mm = mtile * 16 + ((lane < 16) ? r : r + 8);   // §7.12.2 C layout
        stage[gate * 512 + mm * 16 + nn] = acc[r];
      }
    }
    __syncthreads();

    // ---- gate math: 2 cells / thread, c & hsum stay in registers ----
    {
      float cc[2] = {c0, c1};
      float ss[2] = {s0, s1};
      const int cells[2] = {cell0, cell1};
#pragma unroll
      for (int j = 0; j < 2; ++j) {
        const int cell = cells[j];
        float iv = fsig(stage[0 * 512 + cell] + bia[j][0]);
        float fv = fsig(stage[1 * 512 + cell] + bia[j][1]);
        float gv = ftanh(stage[2 * 512 + cell] + bia[j][2]);
        float ov = fsig(stage[3 * 512 + cell] + bia[j][3]);
        float c  = fv * cc[j] + iv * gv;
        float h  = ov * ftanh(c);
        cc[j] = c;
        ss[j] += h;
        const int m = cell >> 4, n = cell & 15;
        hdst[(size_t)m * kH + wg * kNH + n] = (__bf16)h;
      }
      c0 = cc[0]; c1 = cc[1]; s0 = ss[0]; s1 = ss[1];
    }

    // ---- grid-wide step barrier (monotone counter) ----
    __threadfence();
    __syncthreads();
    if (tid == 0) {
      atomicAdd(bar, 1);
      const int target = (t + 1) * (int)gridDim.x;
      while (atomicAdd(bar, 0) < target) __builtin_amdgcn_s_sleep(2);
    }
    __syncthreads();
  }

#if HAVE_TDM
  if (wave == 0) __builtin_amdgcn_s_wait_tensorcnt(0);  // drain final prefetch
#endif

  // ---- emit hsum slice ----
  hsum[(size_t)(cell0 >> 4) * kH + wg * kNH + (cell0 & 15)] = s0;
  hsum[(size_t)(cell1 >> 4) * kH + wg * kNH + (cell1 & 15)] = s1;
}

// ---------------------------------------------------------------- Phase 2 ---
__global__ __launch_bounds__(256)
void k_head(const float* __restrict__ hsum, const float* __restrict__ Wcap,
            const float* __restrict__ bcap, const float* __restrict__ Wout,
            const float* __restrict__ bout, float* __restrict__ out) {
  __shared__ float sv[kD];
  __shared__ float red[256];
  const int b = blockIdx.x, tid = threadIdx.x;

  // u[d] = hsum[b,:] . Wcap[d,:] + T*bcap[d]
  const float* hr = hsum + (size_t)b * kH;
  const float* wr = Wcap + (size_t)tid * kH;
  float u = 0.f;
  for (int k = 0; k < kH; k += 4)
    u += hr[k] * wr[k] + hr[k + 1] * wr[k + 1] +
         hr[k + 2] * wr[k + 2] + hr[k + 3] * wr[k + 3];
  u += (float)kT * bcap[tid];

  red[tid] = u * u;
  __syncthreads();
  for (int s = 128; s > 0; s >>= 1) {
    if (tid < s) red[tid] += red[tid + s];
    __syncthreads();
  }
  const float ns = red[0];
  sv[tid] = u * (ns / (1.0f + ns)) * rsqrtf(ns);
  __syncthreads();

  for (int o = tid; o < kOUT; o += 256) {
    const float* wo = Wout + (size_t)o * kD;
    float acc = 0.f;
    for (int k = 0; k < kD; k += 4)
      acc += sv[k] * wo[k] + sv[k + 1] * wo[k + 1] +
             sv[k + 2] * wo[k + 2] + sv[k + 3] * wo[k + 3];
    out[(size_t)b * kOUT + o] = acc + bout[o];
  }
}

// ----------------------------------------------------------------- launch ---
extern "C" void kernel_launch(void* const* d_in, const int* in_sizes, int n_in,
                              void* d_out, int out_size, void* d_ws,
                              size_t ws_size, hipStream_t stream) {
  const float* x     = (const float*)d_in[0];
  const float* W_ih  = (const float*)d_in[1];
  const float* W_hh  = (const float*)d_in[2];
  const float* b_ih  = (const float*)d_in[3];
  const float* b_hh  = (const float*)d_in[4];
  const float* W_cap = (const float*)d_in[5];
  const float* b_cap = (const float*)d_in[6];
  const float* W_out = (const float*)d_in[7];
  const float* b_out = (const float*)d_in[8];
  float* out = (float*)d_out;
  char*  ws  = (char*)d_ws;

  __bf16* xbf   = (__bf16*)(ws + offXbf);
  __bf16* wpack = (__bf16*)(ws + offWpack);
  __bf16* hbuf  = (__bf16*)(ws + offHbuf);
  float*  hsumb = (float*)(ws + offHsum);
  int*    bar   = (int*)(ws + offBar);

  k_cast_x<<<(kT * kB * kIN) / 256, 256, 0, stream>>>(x, xbf);
  k_pack_w<<<(kNWG * 4 * kNKT * 512) / 256, 256, 0, stream>>>(W_hh, W_ih, wpack);
  k_init<<<(2 * kB * kH) / 256, 256, 0, stream>>>(hbuf, bar);

  k_lstm<<<kNWG, 256, (size_t)kLdsTot, stream>>>(xbf, wpack, b_ih, b_hh,
                                                 hbuf, hsumb, bar);

  k_head<<<kB, 256, 0, stream>>>(hsumb, W_cap, b_cap, W_out, b_out, out);
}
